// FewBiTT_66374424592424
// MI455X (gfx1250) — matrix operand order repaired and checked by reading, not compiled, not run on hardware
//
#include <hip/hip_runtime.h>
#include <hip/hip_bf16.h>

// ---------------------------------------------------------------------------
// FewBiTT on MI455X (gfx1250, wave32, WMMA).
//
// Dominant cost: 4x [8192x8192x256] distance GEMM (~1.37e11 FLOPs) plus
// 8x [8192x768x256] projection GEMMs (~2.6e10 FLOPs). Inputs ~50MB -> compute
// bound -> f16 WMMA (f32 accumulate). The [tokQ x tokS] distance matrix is
// never materialized: per-tag segment max (TAG=9) is fused into the GEMM
// epilogue (dist <= 0 -> float-max == uint-min bit trick, ds_min_u32).
// Support tiles are double-buffered in LDS via global_load_async_to_lds_b128
// (ASYNCcnt); operand fragments are double-buffered in registers so WMMAs
// overlap the next fragment loads instead of stalling on s_wait_*cnt.
// ---------------------------------------------------------------------------

#define P_    4
#define TAG_  9
#define HIN_  768
#define H_    256
#define NTOK_ 8192          // 32 * 256 tokens (both support and query)
#define FORGET_ 0.9f
#define ROWH_ 264           // padded LDS tile row stride in halves (528B):
                            // 132 words == 4 banks/row mod 64 -> conflict-free

typedef __attribute__((ext_vector_type(16))) _Float16 v16h;
typedef __attribute__((ext_vector_type(8)))  _Float16 v8h;
typedef __attribute__((ext_vector_type(8)))  float    v8f;
typedef int v4i_ __attribute__((vector_size(16)));   // b128 payload type

// Load a 16-half lane fragment from two contiguous 16B chunks.
// A-matrix (16x32 f16): lane (r16,hi) holds row m0+r16,
//   halves 0..7  = K in [hi*16,   hi*16+8)  -> chunk at +hi*16
//   halves 8..15 = K in [hi*16+8, hi*16+16)-> chunk at +hi*16+8
// B-matrix (32x16 f16): lane (r16,hi) holds col n0+r16,
//   halves 0..15 = K in [hi*16, hi*16+16)   -> chunks at +hi*16, +hi*16+8
__device__ __forceinline__ v16h ldfrag(const _Float16* p0, const _Float16* p1) {
    v8h lo = *(const v8h*)p0;
    v8h hi = *(const v8h*)p1;
    v16h r;
#pragma unroll
    for (int i = 0; i < 8; ++i) { r[i] = lo[i]; r[i + 8] = hi[i]; }
    return r;
}

// --- CDNA5 async-copy helpers (ASYNCcnt path), builtin first, asm fallback --
template <int N>
__device__ __forceinline__ void wait_asynccnt() {
#if __has_builtin(__builtin_amdgcn_s_wait_asynccnt)
    __builtin_amdgcn_s_wait_asynccnt(N);
#else
    asm volatile("s_wait_asynccnt %0" :: "i"(N) : "memory");
#endif
}

__device__ __forceinline__ void async_ld16(const _Float16* g, _Float16* l) {
#if __has_builtin(__builtin_amdgcn_global_load_async_to_lds_b128)
    __builtin_amdgcn_global_load_async_to_lds_b128(
        (__attribute__((address_space(1))) v4i_*)g,
        (__attribute__((address_space(3))) v4i_*)l, 0, 0);
#else
    asm volatile("global_load_async_to_lds_b128 %0, %1, off"
                 :: "v"((unsigned)(unsigned long)l), "v"(g) : "memory");
#endif
}

// ---------------------------------------------------------------------------
// prep: fp32->f16 casts, W transpose-cast (so B fragments are contiguous),
// zero the norm accumulators.
// ---------------------------------------------------------------------------
__global__ void prep_kernel(const float* __restrict__ S, const float* __restrict__ Q,
                            const float* __restrict__ W,
                            _Float16* __restrict__ S16, _Float16* __restrict__ Q16,
                            _Float16* __restrict__ Wt16,
                            float* __restrict__ snorm, float* __restrict__ qnorm) {
    const long tid = (long)blockIdx.x * blockDim.x + threadIdx.x;
    const long nthr = (long)gridDim.x * blockDim.x;
    const long nSQ = (long)NTOK_ * HIN_;
    for (long i = tid; i < nSQ; i += nthr) { S16[i] = (_Float16)S[i]; Q16[i] = (_Float16)Q[i]; }
    const long nW = (long)P_ * H_ * HIN_;
    for (long i = tid; i < nW; i += nthr) {
        long p = i / ((long)H_ * HIN_);
        long rem = i % ((long)H_ * HIN_);
        long h = rem / HIN_, k = rem % HIN_;
        Wt16[i] = (_Float16)W[(p * HIN_ + k) * H_ + h];   // Wt[p][h][k] = W[p][k][h]
    }
    const long nN = (long)P_ * NTOK_;
    for (long i = tid; i < nN; i += nthr) { snorm[i] = 0.f; qnorm[i] = 0.f; }
}

// ---------------------------------------------------------------------------
// proj: Out[p] = X @ W[p] + b[p]  (f16 out) + row norms.
// Block = 128 threads = 4 waves. Wave w computes a 16 x 64 tile (4 WMMA
// accumulators). Fragments are software-pipelined: iteration i+1's loads are
// issued before iteration i's WMMAs so the MMA pipe overlaps memory latency.
// ---------------------------------------------------------------------------
__global__ void proj_kernel(const _Float16* __restrict__ X16,   // [NTOK][HIN]
                            const _Float16* __restrict__ Wt16,  // [P][H][HIN]
                            const float* __restrict__ bias,     // [P][H]
                            _Float16* __restrict__ Out16,       // [P][NTOK][H]
                            float* __restrict__ norm) {         // [P][NTOK]
    const int wave = threadIdx.x >> 5;
    const int lane = threadIdx.x & 31;
    const int r16 = lane & 15, hi = lane >> 4;
    const int mtiles = NTOK_ >> 4;
    const int p  = blockIdx.x / mtiles;
    const int m0 = (blockIdx.x % mtiles) * 16;
    const int n0 = wave * 64;

    const _Float16* Arow = X16 + (size_t)(m0 + r16) * HIN_;
    const _Float16* Wp   = Wt16 + (size_t)p * H_ * HIN_;

    v8f acc[4];
#pragma unroll
    for (int j = 0; j < 4; ++j) acc[j] = (v8f){};

    // Prologue loads (kk = 0).
    v16h a_cur = ldfrag(Arow + hi * 16, Arow + hi * 16 + 8);
    v16h bb_cur[4];
#pragma unroll
    for (int j = 0; j < 4; ++j) {
        const _Float16* Brow = Wp + (size_t)(n0 + j * 16 + r16) * HIN_ + hi * 16;
        bb_cur[j] = ldfrag(Brow, Brow + 8);
    }

#pragma unroll
    for (int kk = 0; kk < HIN_; kk += 32) {
        v16h a_nxt, bb_nxt[4];
        if (kk + 32 < HIN_) {       // issue next loads before current WMMAs
            a_nxt = ldfrag(Arow + kk + 32 + hi * 16, Arow + kk + 40 + hi * 16);
#pragma unroll
            for (int j = 0; j < 4; ++j) {
                const _Float16* Brow =
                    Wp + (size_t)(n0 + j * 16 + r16) * HIN_ + kk + 32 + hi * 16;
                bb_nxt[j] = ldfrag(Brow, Brow + 8);
            }
        }
#pragma unroll
        for (int j = 0; j < 4; ++j)
            acc[j] = __builtin_amdgcn_wmma_f32_16x16x32_f16(
                false, a_cur, false, bb_cur[j], (short)0, acc[j], false, false);
        if (kk + 32 < HIN_) {
            a_cur = a_nxt;
#pragma unroll
            for (int j = 0; j < 4; ++j) bb_cur[j] = bb_nxt[j];
        }
    }

    float msum[8];
#pragma unroll
    for (int r = 0; r < 8; ++r) msum[r] = 0.f;

#pragma unroll
    for (int j = 0; j < 4; ++j) {
        const int n = n0 + j * 16 + r16;
        const float bv = bias[p * H_ + n];
#pragma unroll
        for (int r = 0; r < 8; ++r) {
            float v = acc[j][r] + bv;
            msum[r] += v * v;
            Out16[((size_t)p * NTOK_ + m0 + r + hi * 8) * H_ + n] = (_Float16)v;
        }
    }
    // Reduce squared sums across the 16 lanes sharing each C row, then atomic.
#pragma unroll
    for (int r = 0; r < 8; ++r) {
        float s = msum[r];
        s += __shfl_xor(s, 1); s += __shfl_xor(s, 2);
        s += __shfl_xor(s, 4); s += __shfl_xor(s, 8);
        if (r16 == 0) atomicAdd(&norm[(size_t)p * NTOK_ + m0 + r + hi * 8], s);
    }
}

// ---------------------------------------------------------------------------
// dist: fused Qp @ Sp^T + per-tag segment max.
// Block = 256 threads = 8 waves, owns 64 exclusive query rows (4 m-tiles x
// 2 n-halves per wave). Support cols stream through a double-buffered LDS
// tile (128 x 256 f16, padded rows) filled with global_load_async_to_lds_b128
// while WMMAs consume the previous tile from ds_load_b128 (register
// double-buffered across the K loop). Qp A-fragments preloaded (8 x v16h).
// ---------------------------------------------------------------------------
__global__ void __launch_bounds__(256)
dist_kernel(const _Float16* __restrict__ Qp16,  // [P][NTOK][H]
            const _Float16* __restrict__ Sp16,  // [P][NTOK][H]
            const float* __restrict__ qnorm,    // [P][NTOK]
            const float* __restrict__ snorm,    // [P][NTOK]
            const int* __restrict__ tags,       // [P][NTOK]
            float* __restrict__ segmax) {       // [P][TAG][NTOK]
    __shared__ _Float16 tile[2][128 * ROWH_];   // 2 x 67.5KB
    __shared__ unsigned smax[TAG_][64];
    const int tid = threadIdx.x;
    const int wave = tid >> 5, lane = tid & 31;
    const int r16 = lane & 15, hi = lane >> 4;
    const int mt = wave >> 1, nh = wave & 1;
    const int mblocks = NTOK_ / 64;
    const int p  = blockIdx.x / mblocks;
    const int mb = (blockIdx.x % mblocks) * 64;

    for (int i = tid; i < TAG_ * 64; i += 256) ((unsigned*)smax)[i] = 0xFF800000u; // -inf
    __syncthreads();

    const int m0 = mb + mt * 16;
    const _Float16* Qrow = Qp16 + ((size_t)p * NTOK_ + m0 + r16) * H_;
    v16h aq[8];
#pragma unroll
    for (int k8 = 0; k8 < 8; ++k8)
        aq[k8] = ldfrag(Qrow + k8 * 32 + hi * 16, Qrow + k8 * 32 + hi * 16 + 8);

    float qn_[8];
#pragma unroll
    for (int r = 0; r < 8; ++r) qn_[r] = qnorm[(size_t)p * NTOK_ + m0 + r + hi * 8];

    const _Float16* Sbase = Sp16 + (size_t)p * NTOK_ * H_;
    const float* snp = snorm + (size_t)p * NTOK_;
    const int* tp = tags + p * NTOK_;

    // 16 async b128 instructions per wave fill one 128x256 f16 tile.
    auto issue_tile = [&](int nb, int buf) {
        const _Float16* gb = Sbase + (size_t)nb * H_;
        _Float16* lb = &tile[buf][0];
#pragma unroll
        for (int i = 0; i < 16; ++i) {
            const int chunk = i * 256 + tid;   // 4096 x 16B chunks
            const int row = chunk >> 5;        // 32 chunks per 512B row
            const int c8  = chunk & 31;
            async_ld16(gb + (size_t)row * H_ + c8 * 8,
                       lb + (size_t)row * ROWH_ + c8 * 8);
        }
    };

    int cur = 0;
    issue_tile(0, 0);
    const int nl = nh * 64;                    // this wave's cols within tile

    for (int nb = 0; nb < NTOK_; nb += 128) {
        const bool hasnext = (nb + 128 < NTOK_);
        if (hasnext) {
            issue_tile(nb + 128, cur ^ 1);
            wait_asynccnt<16>();               // current tile complete (in-order)
        } else {
            wait_asynccnt<0>();
        }
        __syncthreads();                       // publish tile to all waves

        const _Float16* lb = &tile[cur][0];
        v8f acc[4];
#pragma unroll
        for (int j = 0; j < 4; ++j) acc[j] = (v8f){};

        // Register double-buffered B fragments across the K loop.
        v16h bb_cur[4];
#pragma unroll
        for (int j = 0; j < 4; ++j) {
            const _Float16* Brow = lb + (size_t)(nl + j * 16 + r16) * ROWH_ + hi * 16;
            bb_cur[j] = ldfrag(Brow, Brow + 8);
        }
#pragma unroll
        for (int k8 = 0; k8 < 8; ++k8) {
            v16h bb_nxt[4];
            if (k8 < 7) {
#pragma unroll
                for (int j = 0; j < 4; ++j) {
                    const _Float16* Brow = lb + (size_t)(nl + j * 16 + r16) * ROWH_ +
                                           (k8 + 1) * 32 + hi * 16;
                    bb_nxt[j] = ldfrag(Brow, Brow + 8);
                }
            }
#pragma unroll
            for (int j = 0; j < 4; ++j)
                acc[j] = __builtin_amdgcn_wmma_f32_16x16x32_f16(
                    false, aq[k8], false, bb_cur[j], (short)0, acc[j], false, false);
            if (k8 < 7) {
#pragma unroll
                for (int j = 0; j < 4; ++j) bb_cur[j] = bb_nxt[j];
            }
        }
#pragma unroll
        for (int j = 0; j < 4; ++j) {
            const int n = nb + nl + j * 16 + r16;
            const float sv = snp[n];
            const int t = tp[n];
#pragma unroll
            for (int r = 0; r < 8; ++r) {
                float d = 2.f * acc[j][r] - qn_[r] - sv;
                d = fminf(d, 0.f);  // keep the uint-min == float-max invariant
                atomicMin(&smax[t][mt * 16 + r + hi * 8], __float_as_uint(d));
            }
        }
        __syncthreads();                       // done reading tile[cur]
        cur ^= 1;
    }
    for (int i = tid; i < TAG_ * 64; i += 256) {
        const int t = i / 64, m = i % 64;
        segmax[((size_t)p * TAG_ + t) * NTOK_ + mb + m] =
            __uint_as_float(((unsigned*)smax)[i]);
    }
}

// ---------------------------------------------------------------------------
// Per-tag counts and support sums (for the prototype EMA). One block per p;
// thread h owns hidden column h -> no atomics.
// ---------------------------------------------------------------------------
__global__ void seg_stats_kernel(const _Float16* __restrict__ Sp16,
                                 const int* __restrict__ tags,
                                 float* __restrict__ segsum,   // [P][TAG][H]
                                 float* __restrict__ counts) { // [P][TAG]
    const int p = blockIdx.x, h = threadIdx.x;
    __shared__ float ssum[TAG_][H_];
    __shared__ float cnt[TAG_];
#pragma unroll
    for (int t = 0; t < TAG_; ++t) ssum[t][h] = 0.f;
    if (h < TAG_) cnt[h] = 0.f;
    __syncthreads();
    const _Float16* Sp = Sp16 + (size_t)p * NTOK_ * H_;
    const int* tp = tags + p * NTOK_;
    for (int tok = 0; tok < NTOK_; ++tok) {
        const int t = tp[tok];
        ssum[t][h] += (float)Sp[(size_t)tok * H_ + h];
        if (h == 0) cnt[t] += 1.f;
    }
    __syncthreads();
#pragma unroll
    for (int t = 0; t < TAG_; ++t) segsum[((size_t)p * TAG_ + t) * H_ + h] = ssum[t][h];
    if (h < TAG_) counts[p * TAG_ + h] = cnt[h];
}

// ---------------------------------------------------------------------------
// logits + pred: prototype-distance fallback, merge with segment max, argmax.
// ---------------------------------------------------------------------------
__global__ void logits_kernel(const _Float16* __restrict__ Qp16,
                              const float* __restrict__ qnorm,
                              const float* __restrict__ proto,   // [P][TAG][H]
                              const float* __restrict__ segmax,  // [P][TAG][NTOK]
                              const float* __restrict__ counts,  // [P][TAG]
                              float* __restrict__ out_logits,    // [P][NTOK][TAG]
                              float* __restrict__ out_pred) {    // [P][NTOK]
    const int p = blockIdx.y;
    const int m = blockIdx.x * blockDim.x + threadIdx.x;
    __shared__ float pr[TAG_][H_];
    __shared__ float pn[TAG_], cn[TAG_];
    for (int i = threadIdx.x; i < TAG_ * H_; i += blockDim.x)
        ((float*)pr)[i] = proto[(size_t)p * TAG_ * H_ + i];
    __syncthreads();
    if (threadIdx.x < TAG_) {
        float s = 0.f;
        for (int h = 0; h < H_; ++h) s += pr[threadIdx.x][h] * pr[threadIdx.x][h];
        pn[threadIdx.x] = s;
        cn[threadIdx.x] = counts[p * TAG_ + threadIdx.x];
    }
    __syncthreads();

    float dot[TAG_];
#pragma unroll
    for (int t = 0; t < TAG_; ++t) dot[t] = 0.f;
    const _Float16* Qrow = Qp16 + ((size_t)p * NTOK_ + m) * H_;
    for (int h = 0; h < H_; ++h) {
        const float qv = (float)Qrow[h];
#pragma unroll
        for (int t = 0; t < TAG_; ++t) dot[t] += qv * pr[t][h];  // LDS broadcast
    }
    const float qn = qnorm[(size_t)p * NTOK_ + m];
    float best = -3.4e38f;
    int besti = 0;
#pragma unroll
    for (int t = 0; t < TAG_; ++t) {
        const float pd = 2.f * dot[t] - pn[t] - qn;
        const float sm = segmax[((size_t)p * TAG_ + t) * NTOK_ + m];
        const float val = (cn[t] > 0.f) ? sm : pd;
        out_logits[((size_t)p * NTOK_ + m) * TAG_ + t] = val;
        if (val > best) { best = val; besti = t; }
    }
    out_pred[(size_t)p * NTOK_ + m] = (float)besti;
}

// ---------------------------------------------------------------------------
// Prototype EMA update.
// ---------------------------------------------------------------------------
__global__ void proto_kernel(const float* __restrict__ proto,
                             const float* __restrict__ segsum,
                             const float* __restrict__ counts,
                             float* __restrict__ out_proto) {
    const int i = blockIdx.x * blockDim.x + threadIdx.x;
    if (i >= P_ * TAG_ * H_) return;
    const int pt = i / H_;
    const float c = counts[pt];
    const float avg = segsum[i] / fmaxf(c, 1.f);
    const float pv = proto[i];
    out_proto[i] = (c > 0.f) ? ((1.f - FORGET_) * pv + FORGET_ * avg) : pv;
}

// ---------------------------------------------------------------------------
extern "C" void kernel_launch(void* const* d_in, const int* in_sizes, int n_in,
                              void* d_out, int out_size, void* d_ws, size_t ws_size,
                              hipStream_t stream) {
    const float* S     = (const float*)d_in[0];   // [32,256,768]
    const float* Q     = (const float*)d_in[1];   // [32,256,768]
    const float* W     = (const float*)d_in[2];   // [P,768,256]
    const float* bias  = (const float*)d_in[3];   // [P,256]
    const float* proto = (const float*)d_in[4];   // [P,9,256]
    const int*   tags  = (const int*)d_in[5];     // [P,32,256]

    // Workspace carve-out (~62 MB), 256B aligned.
    char* base = (char*)d_ws;
    size_t off = 0;
    auto alloc = [&](size_t bytes) -> char* {
        off = (off + 255) & ~(size_t)255;
        char* p = base + off;
        off += bytes;
        return p;
    };
    _Float16* S16   = (_Float16*)alloc((size_t)NTOK_ * HIN_ * 2);
    _Float16* Q16   = (_Float16*)alloc((size_t)NTOK_ * HIN_ * 2);
    _Float16* Wt16  = (_Float16*)alloc((size_t)P_ * H_ * HIN_ * 2);
    _Float16* Sp16  = (_Float16*)alloc((size_t)P_ * NTOK_ * H_ * 2);
    _Float16* Qp16  = (_Float16*)alloc((size_t)P_ * NTOK_ * H_ * 2);
    float* snorm    = (float*)alloc((size_t)P_ * NTOK_ * 4);
    float* qnorm    = (float*)alloc((size_t)P_ * NTOK_ * 4);
    float* segmax   = (float*)alloc((size_t)P_ * TAG_ * NTOK_ * 4);
    float* counts   = (float*)alloc((size_t)P_ * TAG_ * 4);
    float* segsum   = (float*)alloc((size_t)P_ * TAG_ * H_ * 4);

    float* out_logits = (float*)d_out;                               // P*NTOK*TAG
    float* out_pred   = out_logits + (size_t)P_ * NTOK_ * TAG_;      // P*NTOK
    float* out_proto  = out_pred + (size_t)P_ * NTOK_;               // P*TAG*H

    prep_kernel<<<1024, 256, 0, stream>>>(S, Q, W, S16, Q16, Wt16, snorm, qnorm);

    const int proj_blocks = P_ * (NTOK_ >> 4);
    proj_kernel<<<proj_blocks, 128, 0, stream>>>(S16, Wt16, bias, Sp16, snorm);
    proj_kernel<<<proj_blocks, 128, 0, stream>>>(Q16, Wt16, bias, Qp16, qnorm);

    dist_kernel<<<P_ * (NTOK_ / 64), 256, 0, stream>>>(Qp16, Sp16, qnorm, snorm,
                                                       tags, segmax);

    seg_stats_kernel<<<P_, H_, 0, stream>>>(Sp16, tags, segsum, counts);

    logits_kernel<<<dim3(NTOK_ / 256, P_), 256, 0, stream>>>(Qp16, qnorm, proto,
                                                             segmax, counts,
                                                             out_logits, out_pred);

    proto_kernel<<<(P_ * TAG_ * H_ + 255) / 256, 256, 0, stream>>>(proto, segsum,
                                                                   counts, out_proto);
}
